// Encoder_24910810317610
// MI455X (gfx1250) — matrix operand order repaired and checked by reading, not compiled, not run on hardware
//
#include <hip/hip_runtime.h>

typedef __attribute__((ext_vector_type(16))) _Float16 v16h;
typedef __attribute__((ext_vector_type(8)))  _Float16 v8h;
typedef __attribute__((ext_vector_type(8)))  float    v8f;

namespace {
constexpr int kBatch = 1024;
constexpr int kNW = 300;
constexpr int kN  = 100;   // real tokens
constexpr int kE  = 5;
constexpr int kH  = 4;
constexpr int kD  = 64;
constexpr int kL  = 3;
constexpr int NP  = 128;   // padded tokens: 8 tiles of 16

// LDS strides chosen so 16-lane fragment reads hit distinct banks
constexpr int QK_S = 72;   // f16 per row of Q/K  (row*144B, 16B aligned)
constexpr int VT_S = 136;  // f16 per row of V^T  (row*272B)
constexpr int OC_S = 264;  // f16 per row of concat-head output
constexpr int P_S  = 136;  // f16 per row of per-wave P strip
constexpr int TK_S = 9;    // f32 per token row
constexpr float kNegMax = -3.402823466e38f;
}

__device__ __forceinline__ v16h cat8(v8h lo, v8h hi) {
  return __builtin_shufflevector(lo, hi, 0,1,2,3,4,5,6,7,8,9,10,11,12,13,14,15);
}
// A fragment (16x32 f16): lane holds row M=lane&15; K = 16 contiguous values
// by lane half -> two ds_read_b128, at +0 and +8 elems.
__device__ __forceinline__ v16h frag_a(const _Float16* p) {
  v8h lo = *(const v8h*)p;
  v8h hi = *(const v8h*)(p + 8);
  return cat8(lo, hi);
}
// B fragment (32x16 f16): lane holds col N=lane&15; K = 16 contiguous values by
// lane half -> two adjacent ds_read_b128.
__device__ __forceinline__ v16h frag_b(const _Float16* p) {
  v8h lo = *(const v8h*)p;
  v8h hi = *(const v8h*)(p + 8);
  return cat8(lo, hi);
}
__device__ __forceinline__ float half16_max(float v) {
#pragma unroll
  for (int m = 1; m <= 8; m <<= 1) v = fmaxf(v, __shfl_xor(v, m, 32));
  return v;
}
__device__ __forceinline__ float half16_sum(float v) {
#pragma unroll
  for (int m = 1; m <= 8; m <<= 1) v += __shfl_xor(v, m, 32);
  return v;
}

__global__ __launch_bounds__(128) void encoder_kernel(
    const float* __restrict__ x,
    const float* __restrict__ Wq, const float* __restrict__ Wk,
    const float* __restrict__ Wv, const float* __restrict__ Wo,
    const float* __restrict__ Wf, const float* __restrict__ bfv,
    const float* __restrict__ g1, const float* __restrict__ b1,
    const float* __restrict__ g2, const float* __restrict__ b2,
    float* __restrict__ out) {
  __shared__ __align__(16) float     tok[NP * TK_S];
  __shared__ __align__(16) float     maskv[NP];
  __shared__ __align__(16) float     WoS[kE * kH * kD];      // 5x256 layer slice
  __shared__ __align__(16) _Float16  Qs[NP * QK_S];
  __shared__ __align__(16) _Float16  Ks[NP * QK_S];
  __shared__ __align__(16) _Float16  Vt[kD * VT_S];
  __shared__ __align__(16) _Float16  Oc[NP * OC_S];
  __shared__ __align__(16) _Float16  Pb[4 * 16 * P_S];

  const int tid  = threadIdx.x;
  const int lane = tid & 31;
  const int wave = tid >> 5;
  const int half = lane >> 4;   // lane half within wave32
  const int lm   = lane & 15;
  const int b    = blockIdx.x;

  // ---- token init: reshape to [N][3], mask = conf>0, cat PE(sin,cos), zero masked ----
  for (int n = tid; n < NP; n += 128) {
    float m = 0.f, t0 = 0.f, t1 = 0.f, t2 = 0.f, t3 = 0.f, t4 = 0.f;
    if (n < kN) {
      float k0 = x[b * kNW + n * 3 + 0];
      float k1 = x[b * kNW + n * 3 + 1];
      float k2 = x[b * kNW + n * 3 + 2];
      m = (k2 > 0.f) ? 1.f : 0.f;
      if (m != 0.f) {
        t0 = k0; t1 = k1; t2 = k2;
        t3 = __sinf((float)n); t4 = __cosf((float)n);
      }
    }
    maskv[n] = m;
    tok[n * TK_S + 0] = t0; tok[n * TK_S + 1] = t1; tok[n * TK_S + 2] = t2;
    tok[n * TK_S + 3] = t3; tok[n * TK_S + 4] = t4;
    tok[n * TK_S + 5] = 0.f; tok[n * TK_S + 6] = 0.f; tok[n * TK_S + 7] = 0.f;
  }
  __syncthreads();

  for (int l = 0; l < kL; ++l) {
    // cache this layer's Wo slice (5x256) in LDS
    for (int i = tid; i < kE * kH * kD; i += 128) WoS[i] = Wo[l * kE * kH * kD + i];

    for (int h = 0; h < kH; ++h) {
      // ---- QKV projection (E=5 inner): thread owns fixed d, streams tokens ----
      {
        const int d = tid & 63;
        const float* wqp = Wq + (((l * kH + h) * kD) + d) * kE;
        const float* wkp = Wk + (((l * kH + h) * kD) + d) * kE;
        const float* wvp = Wv + (((l * kH + h) * kD) + d) * kE;
        float q0 = wqp[0], q1 = wqp[1], q2 = wqp[2], q3 = wqp[3], q4 = wqp[4];
        float k0 = wkp[0], k1 = wkp[1], k2 = wkp[2], k3 = wkp[3], k4 = wkp[4];
        float v0 = wvp[0], v1 = wvp[1], v2 = wvp[2], v3 = wvp[3], v4 = wvp[4];
        for (int n = (tid >> 6); n < NP; n += 2) {
          float a0 = tok[n * TK_S + 0], a1 = tok[n * TK_S + 1],
                a2 = tok[n * TK_S + 2], a3 = tok[n * TK_S + 3],
                a4 = tok[n * TK_S + 4];
          float q = a0 * q0 + a1 * q1 + a2 * q2 + a3 * q3 + a4 * q4;
          float k = a0 * k0 + a1 * k1 + a2 * k2 + a3 * k3 + a4 * k4;
          float v = a0 * v0 + a1 * v1 + a2 * v2 + a3 * v3 + a4 * v4;
          Qs[n * QK_S + d] = (_Float16)(q * 0.125f);   // 1/sqrt(64) folded in
          Ks[n * QK_S + d] = (_Float16)k;
          Vt[d * VT_S + n] = (_Float16)v;              // V stored transposed
        }
      }
      __syncthreads();

      // ---- attention: wave handles row strips {wave, wave+4} ----
      for (int sp = wave; sp < 8; sp += 4) {
        const int row = sp * 16 + lm;                  // A-matrix row (query idx)
        v8f acc[8];
#pragma unroll
        for (int t = 0; t < 8; ++t) acc[t] = (v8f){0,0,0,0,0,0,0,0};

        // S = (Q*scale) K^T : 2 K-steps x 8 column tiles, software-pipelined:
        // fragment i+1 is loaded (into fresh VGPRs, live across the WMMA)
        // before WMMA i issues, so DS latency overlaps the matrix pipe.
        {
          const _Float16* aQ = &Qs[row * QK_S + half * 16];
          v16h af = frag_a(aQ);                                   // kb = 0
          v16h bf_ = frag_b(&Ks[lm * QK_S + half * 16]);          // kb=0, t=0
#pragma unroll
          for (int i = 0; i < 16; ++i) {
            const int t = i & 7;
            v16h bnx, anx;
            if (i < 15) {
              const int kb2 = (i + 1) >> 3, t2 = (i + 1) & 7;
              bnx = frag_b(&Ks[(t2 * 16 + lm) * QK_S + kb2 * 32 + half * 16]);
            }
            if (i == 7) anx = frag_a(aQ + 32);                    // kb = 1
            acc[t] = __builtin_amdgcn_wmma_f32_16x16x32_f16(
                false, af, false, bf_, (short)0, acc[t], false, false);
            if (i < 15) bf_ = bnx;
            if (i == 7) af = anx;
          }
        }

        // ---- mask + row softmax in registers (C layout: lane=col, vgpr=row) ----
        float maskj[8];
#pragma unroll
        for (int t = 0; t < 8; ++t) {
          int j = t * 16 + lm;
          maskj[t] = (j < kN) ? maskv[j] : -1.f;       // -1 marks padded column
        }
#pragma unroll
        for (int r = 0; r < 8; ++r) {
          const int i = sp * 16 + r + half * 8;
          const float mi = maskv[i];
          float mx = -__builtin_inff();
#pragma unroll
          for (int t = 0; t < 8; ++t) {
            float s = acc[t][r];
            if (maskj[t] < 0.f)                 s = -__builtin_inff();  // pad col
            else if (mi == 0.f || maskj[t] == 0.f) s = kNegMax;         // pair mask
            acc[t][r] = s;
            mx = fmaxf(mx, s);
          }
          mx = half16_max(mx);
          float sum = 0.f;
#pragma unroll
          for (int t = 0; t < 8; ++t) {
            float p = __expf(acc[t][r] - mx);
            acc[t][r] = p;
            sum += p;
          }
          sum = half16_sum(sum);
          const float inv = 1.f / sum;
#pragma unroll
          for (int t = 0; t < 8; ++t) acc[t][r] *= inv;
        }

        // ---- stage P strip to LDS (row-major f16) for A-fragment reload ----
        _Float16* pw = &Pb[wave * 16 * P_S];
#pragma unroll
        for (int t = 0; t < 8; ++t)
#pragma unroll
          for (int r = 0; r < 8; ++r)
            pw[(r + half * 8) * P_S + t * 16 + lm] = (_Float16)acc[t][r];

        // ---- O = P V : 4 K-steps x 4 column tiles, same rotated pipeline ----
        v8f oacc[4];
#pragma unroll
        for (int t = 0; t < 4; ++t) oacc[t] = (v8f){0,0,0,0,0,0,0,0};
        {
          const _Float16* aP = &pw[lm * P_S + half * 16];
          v16h af = frag_a(aP);                                   // kb = 0
          v16h bf_ = frag_b(&Vt[lm * VT_S + half * 16]);          // kb=0, t=0
#pragma unroll
          for (int i = 0; i < 16; ++i) {
            const int kb = i >> 2, t = i & 3;
            v16h bnx, anx;
            if (i < 15) {
              const int kb2 = (i + 1) >> 2, t2 = (i + 1) & 3;
              bnx = frag_b(&Vt[(t2 * 16 + lm) * VT_S + kb2 * 32 + half * 16]);
            }
            if (t == 3 && kb < 3) anx = frag_a(aP + (kb + 1) * 32);
            oacc[t] = __builtin_amdgcn_wmma_f32_16x16x32_f16(
                false, af, false, bf_, (short)0, oacc[t], false, false);
            if (i < 15) bf_ = bnx;
            if (t == 3 && kb < 3) af = anx;
          }
        }
#pragma unroll
        for (int t = 0; t < 4; ++t)
#pragma unroll
          for (int r = 0; r < 8; ++r)
            Oc[(sp * 16 + r + half * 8) * OC_S + h * kD + t * 16 + lm] =
                (_Float16)oacc[t][r];
      }
      __syncthreads();   // Oc ready / Q,K,Vt free for next head
    }

    // ---- per-token epilogue: Wo proj + res + LN1 + FFN + res + LN2 ----
    if (tid < kN) {
      const int n = tid;
      float ao[5] = {0.f, 0.f, 0.f, 0.f, 0.f};
      const _Float16* ocr = &Oc[n * OC_S];
      // vectorized: 32 x ds_read_b128 instead of 256 x ds_read_u16
      for (int c8 = 0; c8 < (kH * kD) / 8; ++c8) {
        v8h ov8 = *(const v8h*)(ocr + c8 * 8);
#pragma unroll
        for (int u = 0; u < 8; ++u) {
          float ov = (float)ov8[u];
          const int c = c8 * 8 + u;
#pragma unroll
          for (int e = 0; e < 5; ++e) ao[e] += ov * WoS[e * (kH * kD) + c];
        }
      }
      float r1[5], ln1[5];
      float mu = 0.f;
#pragma unroll
      for (int e = 0; e < 5; ++e) { r1[e] = ao[e] + tok[n * TK_S + e]; mu += r1[e]; }
      mu *= 0.2f;
      float var = 0.f;
#pragma unroll
      for (int e = 0; e < 5; ++e) { float d = r1[e] - mu; var += d * d; }
      var *= 0.2f;
      float rstd = rsqrtf(var + 1e-5f);
#pragma unroll
      for (int e = 0; e < 5; ++e)
        ln1[e] = (r1[e] - mu) * rstd * g1[l * 5 + e] + b1[l * 5 + e];
      float r2[5];
      float mu2 = 0.f;
#pragma unroll
      for (int e = 0; e < 5; ++e) {
        float s = bfv[l * 5 + e];
#pragma unroll
        for (int e2 = 0; e2 < 5; ++e2) s += ln1[e2] * Wf[(l * 5 + e) * 5 + e2];
        s = fmaxf(s, 0.f);          // ReLU
        r2[e] = s + ln1[e];
        mu2 += r2[e];
      }
      mu2 *= 0.2f;
      float var2 = 0.f;
#pragma unroll
      for (int e = 0; e < 5; ++e) { float d = r2[e] - mu2; var2 += d * d; }
      var2 *= 0.2f;
      float rstd2 = rsqrtf(var2 + 1e-5f);
#pragma unroll
      for (int e = 0; e < 5; ++e)
        tok[n * TK_S + e] =
            (r2[e] - mu2) * rstd2 * g2[l * 5 + e] + b2[l * 5 + e];
    }
    // warm next layer's projection weights in L2 (global_prefetch_b8)
    if (l + 1 < kL) {
      __builtin_prefetch(Wq + (l + 1) * kH * kD * kE + tid * 8, 0, 0);
      __builtin_prefetch(Wk + (l + 1) * kH * kD * kE + tid * 8, 0, 0);
      __builtin_prefetch(Wv + (l + 1) * kH * kD * kE + tid * 8, 0, 0);
    }
    __syncthreads();
  }

  // ---- final store [B,N,E] ----
  if (tid < kN) {
#pragma unroll
    for (int e = 0; e < 5; ++e)
      out[b * kN * kE + tid * kE + e] = tok[tid * TK_S + e];
  }
}

extern "C" void kernel_launch(void* const* d_in, const int* in_sizes, int n_in,
                              void* d_out, int out_size, void* d_ws, size_t ws_size,
                              hipStream_t stream) {
  (void)in_sizes; (void)n_in; (void)out_size; (void)d_ws; (void)ws_size;
  const float* x  = (const float*)d_in[0];
  const float* Wq = (const float*)d_in[1];
  const float* Wk = (const float*)d_in[2];
  const float* Wv = (const float*)d_in[3];
  const float* Wo = (const float*)d_in[4];
  const float* Wf = (const float*)d_in[5];
  const float* bf = (const float*)d_in[6];
  const float* g1 = (const float*)d_in[7];
  const float* b1 = (const float*)d_in[8];
  const float* g2 = (const float*)d_in[9];
  const float* b2 = (const float*)d_in[10];
  float* out = (float*)d_out;
  encoder_kernel<<<dim3(kBatch), dim3(128), 0, stream>>>(
      x, Wq, Wk, Wv, Wo, Wf, bf, g1, b1, g2, b2, out);
}
